// CrossWindowAttentionBlock_24094766530837
// MI455X (gfx1250) — compile-verified, matrix operand order checked by reading, not run on hardware
//
#include <hip/hip_runtime.h>
#include <math.h>

typedef __attribute__((ext_vector_type(16))) __bf16 v16bf;
typedef __attribute__((ext_vector_type(8)))  float  v8f;
typedef __attribute__((ext_vector_type(4)))  unsigned u32x4;

#define B_     8
#define CIN    96
#define HW_    128
#define C_     192
#define T_     131072      /* B_*HW_*HW_ */
#define KCONV  864         /* 9*96 */
#define SCALE_ 0.2041241452319315f
#define EPS_   1e-5f

__device__ __forceinline__ __bf16 f2bf(float f){
  unsigned u = __builtin_bit_cast(unsigned, f);
  unsigned r = u + 0x7FFFu + ((u >> 16) & 1u);
  unsigned short h = (unsigned short)(r >> 16);
  return __builtin_bit_cast(__bf16, h);
}

__device__ __forceinline__ v8f zf8(){
  v8f z;
#pragma unroll
  for (int i = 0; i < 8; i++) z[i] = 0.f;
  return z;
}

// Gather a 16x32 bf16 WMMA A/B fragment from row-major memory (stride in elems,
// multiple of 8). For A: rows = M. For B: pass B^T (N x K row-major).
__device__ __forceinline__ v16bf load_frag(const __bf16* base, int stride, int k0){
  int lane = threadIdx.x & 31;
  int r = lane & 15, hf = lane >> 4;
  const __bf16* p = base + (size_t)r * stride + k0 + hf * 8;
  union { v16bf v; u32x4 q[2]; } u;
  u.q[0] = *(const u32x4*)(p);
  u.q[1] = *(const u32x4*)(p + 16);
  return u.v;
}

__device__ __forceinline__ v8f wmma_bf16(v16bf a, v16bf b, v8f c){
  return __builtin_amdgcn_wmma_f32_16x16x32_bf16(false, a, false, b, (short)0, c, false, false);
}

__device__ __forceinline__ float wred(float x){
#pragma unroll
  for (int o = 16; o; o >>= 1) x += __shfl_xor(x, o, 32);
  return x;
}

// ---------------- prep kernels ----------------
__global__ void k_zero(u32x4* p, int n){
  u32x4 z = {0u, 0u, 0u, 0u};
  for (int i = blockIdx.x * blockDim.x + threadIdx.x; i < n; i += gridDim.x * blockDim.x)
    p[i] = z;
}

__global__ void k_cast(const float* __restrict__ s, __bf16* __restrict__ d, int n){
  for (int i = blockIdx.x * blockDim.x + threadIdx.x; i < n; i += gridDim.x * blockDim.x)
    d[i] = f2bf(s[i]);
}

// OIHW (192,96,3,3) -> [n][t*96+cin] bf16, t = dy*3+dx
__global__ void k_repack_conv(const float* __restrict__ w, __bf16* __restrict__ rep){
  int idx = blockIdx.x * 256 + threadIdx.x;
  if (idx >= 192 * KCONV) return;
  int n = idx / KCONV, k = idx - n * KCONV;
  int t = k / CIN, cin = k - t * CIN;
  int dy = t / 3, dx = t % 3;
  rep[idx] = f2bf(w[((n * CIN + cin) * 3 + dy) * 3 + dx]);
}

// relative position bias table -> dense [h][64][64] f32
__global__ void k_biasfull(const float* __restrict__ rpb, float* __restrict__ bf){
  int idx = blockIdx.x * 256 + threadIdx.x;
  if (idx >= 8 * 64 * 64) return;
  int h = idx >> 12, i = (idx >> 6) & 63, j = idx & 63;
  int iy = i >> 3, ix = i & 7, jy = j >> 3, jx = j & 7;
  int rel = (iy - jy + 7) * 15 + (ix - jx + 7);
  bf[idx] = rpb[rel * 8 + h];
}

// ---------------- conv3x3 via implicit-GEMM WMMA ----------------
__global__ void __launch_bounds__(192) k_conv3(const float* __restrict__ src,
        const __bf16* __restrict__ wrep, const float* __restrict__ bias,
        float* __restrict__ dst /* NHWC f32 */){
  __shared__ __align__(16) __bf16 At[16 * KCONV];
  int blk = blockIdx.x;
  int b = blk >> 10, rem = blk & 1023;
  int y = rem >> 3, x0 = (rem & 7) << 4;
  for (int idx = threadIdx.x; idx < 16 * KCONV; idx += 192){
    int p = idx / KCONV, kk = idx - p * KCONV;
    int t = kk / CIN, cin = kk - t * CIN;
    int gy = y + t / 3 - 1, gx = x0 + p + t % 3 - 1;
    float v = 0.f;
    if (gy >= 0 && gy < HW_ && gx >= 0 && gx < HW_)
      v = src[((b * CIN + cin) * HW_ + gy) * HW_ + gx];
    At[p * KCONV + kk] = f2bf(v);
  }
  __syncthreads();
  int wv = threadIdx.x >> 5, lane = threadIdx.x & 31;
  int col = lane & 15, hf = lane >> 4;
  int n0 = wv * 32;
  v8f a0 = zf8(), a1 = zf8();
  for (int s = 0; s < 27; ++s){
    v16bf af = load_frag(At, KCONV, s * 32);
    v16bf b0 = load_frag(wrep + (size_t)n0 * KCONV, KCONV, s * 32);
    a0 = wmma_bf16(af, b0, a0);
    v16bf b1 = load_frag(wrep + (size_t)(n0 + 16) * KCONV, KCONV, s * 32);
    a1 = wmma_bf16(af, b1, a1);
  }
  size_t tokbase = (size_t)((b * HW_ + y) * HW_ + x0);
  int c0 = n0 + col, c1 = c0 + 16;
  float bi0 = bias[c0], bi1 = bias[c1];
#pragma unroll
  for (int j = 0; j < 8; j++){
    int row = j + 8 * hf;
    dst[(tokbase + row) * C_ + c0] = a0[j] + bi0;
    dst[(tokbase + row) * C_ + c1] = a1[j] + bi1;
  }
}

// ---------------- LayerNorm + window partition (f32 NHWC -> bf16 windowed) ----------------
__global__ void __launch_bounds__(256) k_ln_win(const float* __restrict__ in,
        const float* __restrict__ gw, const float* __restrict__ gb,
        __bf16* __restrict__ out){
  int tok = blockIdx.x * 8 + (threadIdx.x >> 5);
  int lane = threadIdx.x & 31;
  const float* p = in + (size_t)tok * C_;
  float v[6], s = 0.f, ss = 0.f;
#pragma unroll
  for (int i = 0; i < 6; i++){ v[i] = p[lane * 6 + i]; s += v[i]; ss += v[i] * v[i]; }
  s = wred(s); ss = wred(ss);
  float mean = s * (1.f / C_);
  float rs = rsqrtf(ss * (1.f / C_) - mean * mean + EPS_);
  int b = tok >> 14, y = (tok >> 7) & 127, x = tok & 127;
  int w = b * 256 + (y >> 3) * 16 + (x >> 3);
  int tl = (y & 7) * 8 + (x & 7);
  __bf16* q = out + (size_t)(w * 64 + tl) * C_;
#pragma unroll
  for (int i = 0; i < 6; i++){
    int c = lane * 6 + i;
    q[c] = f2bf((v[i] - mean) * rs * gw[c] + gb[c]);
  }
}

// ---------------- generic WMMA GEMM: C = A(MxK) * Bw(NxK)^T, epilogue by mode ----------------
// mode 0: Q  -> bf16, head-padded layout [tok][h*32+d], * scale
// mode 1: K  -> bf16, same layout
// mode 2: V  -> bf16, transposed [ (w*8+h)*32+d ][ tl ]
// mode 3: AP -> f32 [tok][192], + bias
__global__ void __launch_bounds__(128) k_gemm(const __bf16* __restrict__ A, int lda,
        const __bf16* __restrict__ Bw, int ldb, const float* __restrict__ bias,
        void* __restrict__ outp, int K, float scale, int mode){
  int m0 = blockIdx.x * 16;
  int wv = threadIdx.x >> 5;
  int n0 = (blockIdx.y * 4 + wv) * 16;
  int lane = threadIdx.x & 31;
  int col = lane & 15, hf = lane >> 4;
  v8f acc = zf8();
  for (int ks = 0; ks * 32 < K; ++ks){
    v16bf af = load_frag(A + (size_t)m0 * lda, lda, ks * 32);
    v16bf bf = load_frag(Bw + (size_t)n0 * ldb, ldb, ks * 32);
    acc = wmma_bf16(af, bf, acc);
  }
  if (mode <= 1){
    __bf16* o = (__bf16*)outp;
    int c = n0 + col; int h = c / 24, d = c - h * 24;
#pragma unroll
    for (int j = 0; j < 8; j++){
      int row = j + 8 * hf;
      o[(size_t)(m0 + row) * 256 + h * 32 + d] = f2bf(acc[j] * scale);
    }
  } else if (mode == 2){
    __bf16* o = (__bf16*)outp;
    int c = n0 + col; int h = c / 24, d = c - h * 24;
#pragma unroll
    for (int j = 0; j < 8; j++){
      int tok = m0 + j + 8 * hf;
      int w = tok >> 6, tl = tok & 63;
      o[((size_t)((w * 8 + h) * 32 + d)) * 64 + tl] = f2bf(acc[j]);
    }
  } else {
    float* o = (float*)outp;
    int c = n0 + col;
    float bi = bias[c];
#pragma unroll
    for (int j = 0; j < 8; j++){
      int row = j + 8 * hf;
      o[(size_t)(m0 + row) * C_ + c] = acc[j] + bi;
    }
  }
}

// ---------------- attention per (window, head) ----------------
__global__ void __launch_bounds__(32) k_attn(const __bf16* __restrict__ qb,
        const __bf16* __restrict__ kb, const __bf16* __restrict__ vT,
        const float* __restrict__ biasF, __bf16* __restrict__ oa){
  __shared__ __align__(16) float S[64 * 64];
  __shared__ __align__(16) __bf16 P[64 * 72];
  int w = blockIdx.x >> 3, h = blockIdx.x & 7;
  int lane = threadIdx.x & 31;
  int col = lane & 15, hf = lane >> 4;
  size_t qkbase = (size_t)w * 64 * 256 + h * 32;
  for (int mt = 0; mt < 4; ++mt){
    v16bf af = load_frag(qb + qkbase + (size_t)mt * 16 * 256, 256, 0);
    for (int nt = 0; nt < 4; ++nt){
      v16bf bf = load_frag(kb + qkbase + (size_t)nt * 16 * 256, 256, 0);
      v8f acc = wmma_bf16(af, bf, zf8());
#pragma unroll
      for (int j = 0; j < 8; j++){
        int qi = mt * 16 + j + 8 * hf;
        int kj = nt * 16 + col;
        S[qi * 64 + kj] = acc[j] + biasF[(h * 64 + qi) * 64 + kj];
      }
    }
  }
  __syncthreads();
  for (int rr = lane; rr < 64; rr += 32){
    float mx = -3.4e38f;
    for (int c2 = 0; c2 < 64; c2++) mx = fmaxf(mx, S[rr * 64 + c2]);
    float sm = 0.f;
    for (int c2 = 0; c2 < 64; c2++) sm += expf(S[rr * 64 + c2] - mx);
    float inv = 1.f / sm;
    for (int c2 = 0; c2 < 64; c2++) P[rr * 72 + c2] = f2bf(expf(S[rr * 64 + c2] - mx) * inv);
  }
  __syncthreads();
  for (int mt = 0; mt < 4; ++mt){
    for (int nt = 0; nt < 2; ++nt){
      v8f acc = zf8();
#pragma unroll
      for (int ks = 0; ks < 2; ++ks){
        v16bf af = load_frag(P + mt * 16 * 72, 72, ks * 32);
        v16bf bf = load_frag(vT + ((size_t)(w * 8 + h) * 32 + nt * 16) * 64, 64, ks * 32);
        acc = wmma_bf16(af, bf, acc);
      }
      int dd = nt * 16 + col;
      if (dd < 24){
#pragma unroll
        for (int j = 0; j < 8; j++){
          int qi = mt * 16 + j + 8 * hf;
          oa[(size_t)(w * 64 + qi) * C_ + h * 24 + dd] = f2bf(acc[j]);
        }
      }
    }
  }
}

// ---------------- fused LN3 -> fc1 -> gelu -> fc2 -> x2 -> LN4 -> +residuals, NCHW ----------------
__global__ void __launch_bounds__(256) k_mlp(const float* __restrict__ xin,
        const float* __restrict__ n3w, const float* __restrict__ n3b,
        const __bf16* __restrict__ w1, const float* __restrict__ b1,
        const __bf16* __restrict__ w2, const float* __restrict__ b2,
        const float* __restrict__ n4w, const float* __restrict__ n4b,
        const float* __restrict__ xp, const float* __restrict__ vp,
        float* __restrict__ dout){
  __shared__ __align__(16) __bf16 Al[16 * 200];
  __shared__ __align__(16) __bf16 Hl[16 * 776];
  __shared__ __align__(16) float  Ml[16 * 192];
  int tw0 = blockIdx.x * 16;
  int wv = threadIdx.x >> 5, lane = threadIdx.x & 31;
  int col = lane & 15, hf = lane >> 4;
  // LN3 into bf16 A tile
  for (int rr = wv * 2; rr < wv * 2 + 2; ++rr){
    const float* p = xin + (size_t)(tw0 + rr) * C_;
    float v[6], s = 0.f, ss = 0.f;
#pragma unroll
    for (int i = 0; i < 6; i++){ v[i] = p[lane * 6 + i]; s += v[i]; ss += v[i] * v[i]; }
    s = wred(s); ss = wred(ss);
    float mean = s * (1.f / C_);
    float rs = rsqrtf(ss * (1.f / C_) - mean * mean + EPS_);
#pragma unroll
    for (int i = 0; i < 6; i++){
      int c = lane * 6 + i;
      Al[rr * 200 + c] = f2bf((v[i] - mean) * rs * n3w[c] + n3b[c]);
    }
  }
  __syncthreads();
  // fc1 + exact GELU
  for (int t = 0; t < 6; t++){
    int n0 = (wv * 6 + t) * 16;
    v8f acc = zf8();
#pragma unroll
    for (int ks = 0; ks < 6; ks++){
      v16bf af = load_frag(Al, 200, ks * 32);
      v16bf bf = load_frag(w1 + (size_t)n0 * C_, C_, ks * 32);
      acc = wmma_bf16(af, bf, acc);
    }
    int c = n0 + col; float bi = b1[c];
#pragma unroll
    for (int j = 0; j < 8; j++){
      int row = j + 8 * hf;
      float xv = acc[j] + bi;
      float g = 0.5f * xv * (1.f + erff(xv * 0.70710678118f));
      Hl[row * 776 + c] = f2bf(g);
    }
  }
  __syncthreads();
  // fc2, m = 2*(h@W2^T + b2)
  for (int nt = wv; nt < 12; nt += 8){
    int n0 = nt * 16;
    v8f acc = zf8();
    for (int ks = 0; ks < 24; ks++){
      v16bf af = load_frag(Hl, 776, ks * 32);
      v16bf bf = load_frag(w2 + (size_t)n0 * 768, 768, ks * 32);
      acc = wmma_bf16(af, bf, acc);
    }
    int c = n0 + col; float bi = b2[c];
#pragma unroll
    for (int j = 0; j < 8; j++){
      int row = j + 8 * hf;
      Ml[row * 192 + c] = 2.f * (acc[j] + bi);
    }
  }
  __syncthreads();
  // LN4 + residual adds, window-reverse, NCHW store
  for (int rr = wv * 2; rr < wv * 2 + 2; ++rr){
    float v[6], s = 0.f, ss = 0.f;
#pragma unroll
    for (int i = 0; i < 6; i++){ v[i] = Ml[rr * 192 + lane * 6 + i]; s += v[i]; ss += v[i] * v[i]; }
    s = wred(s); ss = wred(ss);
    float mean = s * (1.f / C_);
    float rs = rsqrtf(ss * (1.f / C_) - mean * mean + EPS_);
    int tok = tw0 + rr;
    int w = tok >> 6, tl = tok & 63;
    int b = w >> 8, wy = (w >> 4) & 15, wx = w & 15;
    int y = wy * 8 + (tl >> 3), x = wx * 8 + (tl & 7);
    size_t rtok = (size_t)((b * HW_ + y) * HW_ + x);
#pragma unroll
    for (int i = 0; i < 6; i++){
      int c = lane * 6 + i;
      float o = (v[i] - mean) * rs * n4w[c] + n4b[c] + xp[rtok * C_ + c] + vp[rtok * C_ + c];
      dout[(((size_t)b * C_ + c) * HW_ + y) * HW_ + x] = o;
    }
  }
}

extern "C" void kernel_launch(void* const* d_in, const int* in_sizes, int n_in,
                              void* d_out, int out_size, void* d_ws, size_t ws_size,
                              hipStream_t stream) {
  (void)in_sizes; (void)n_in; (void)out_size; (void)ws_size;
  const float* x     = (const float*)d_in[0];
  const float* v     = (const float*)d_in[1];
  const float* pq_w  = (const float*)d_in[2];
  const float* pq_b  = (const float*)d_in[3];
  const float* pv_w  = (const float*)d_in[4];
  const float* pv_b  = (const float*)d_in[5];
  const float* n1w   = (const float*)d_in[6];
  const float* n1b   = (const float*)d_in[7];
  const float* n2w   = (const float*)d_in[8];
  const float* n2b   = (const float*)d_in[9];
  const float* n3w   = (const float*)d_in[10];
  const float* n3b   = (const float*)d_in[11];
  const float* n4w   = (const float*)d_in[12];
  const float* n4b   = (const float*)d_in[13];
  const float* q_w   = (const float*)d_in[14];
  const float* kv_w  = (const float*)d_in[15];
  const float* ap_w  = (const float*)d_in[16];
  const float* ap_b  = (const float*)d_in[17];
  const float* rpb   = (const float*)d_in[18];
  const float* fc1_w = (const float*)d_in[19];
  const float* fc1_b = (const float*)d_in[20];
  const float* fc2_w = (const float*)d_in[21];
  const float* fc2_b = (const float*)d_in[22];

  char* ws = (char*)d_ws;
  const size_t szF = (size_t)T_ * C_ * 4;   // 100663296
  const size_t szH = (size_t)T_ * C_ * 2;   // 50331648
  const size_t szP = (size_t)T_ * 256 * 2;  // 67108864
  size_t o_xp  = 0;
  size_t o_vp  = o_xp + szF;
  size_t o_xln = o_vp + szF;
  size_t o_vln = o_xln + szH;
  size_t o_q   = o_vln + szH;
  size_t o_k   = o_q + szP;
  size_t o_vT  = o_k + szP;
  size_t o_oa  = o_vT + szP;
  size_t o_w   = o_oa + szH;
  size_t o_wq  = o_w;                 // conv-x weights repacked bf16 (192*864)
  size_t o_wv  = o_wq + 192 * KCONV * 2;
  size_t o_qw  = o_wv + 192 * KCONV * 2;
  size_t o_kvw = o_qw + 192 * 192 * 2;
  size_t o_apw = o_kvw + 384 * 192 * 2;
  size_t o_f1  = o_apw + 192 * 192 * 2;
  size_t o_f2  = o_f1 + 768 * 192 * 2;
  size_t o_bf  = o_f2 + 192 * 768 * 2;
  size_t o_op  = o_xln;               // f32 out_proj aliases consumed xln+vln

  float*  xp   = (float*)(ws + o_xp);
  float*  vp   = (float*)(ws + o_vp);
  __bf16* xln  = (__bf16*)(ws + o_xln);
  __bf16* vln  = (__bf16*)(ws + o_vln);
  __bf16* qbuf = (__bf16*)(ws + o_q);
  __bf16* kbuf = (__bf16*)(ws + o_k);
  __bf16* vT   = (__bf16*)(ws + o_vT);
  __bf16* oa   = (__bf16*)(ws + o_oa);
  __bf16* wqr  = (__bf16*)(ws + o_wq);
  __bf16* wvr  = (__bf16*)(ws + o_wv);
  __bf16* qwb  = (__bf16*)(ws + o_qw);
  __bf16* kvwb = (__bf16*)(ws + o_kvw);
  __bf16* apwb = (__bf16*)(ws + o_apw);
  __bf16* f1b  = (__bf16*)(ws + o_f1);
  __bf16* f2b  = (__bf16*)(ws + o_f2);
  float*  bF   = (float*)(ws + o_bf);
  float*  oproj= (float*)(ws + o_op);

  // zero q/k/vT (covers the HD 24->32 pad lanes)
  k_zero<<<4096, 256, 0, stream>>>((u32x4*)(ws + o_q), (int)(3 * szP / 16));
  // weight prep
  k_repack_conv<<<(192 * KCONV + 255) / 256, 256, 0, stream>>>(pq_w, wqr);
  k_repack_conv<<<(192 * KCONV + 255) / 256, 256, 0, stream>>>(pv_w, wvr);
  k_cast<<<64, 256, 0, stream>>>(q_w,   qwb,  192 * 192);
  k_cast<<<64, 256, 0, stream>>>(kv_w,  kvwb, 384 * 192);
  k_cast<<<64, 256, 0, stream>>>(ap_w,  apwb, 192 * 192);
  k_cast<<<128, 256, 0, stream>>>(fc1_w, f1b,  768 * 192);
  k_cast<<<128, 256, 0, stream>>>(fc2_w, f2b,  192 * 768);
  k_biasfull<<<128, 256, 0, stream>>>(rpb, bF);

  // conv projections -> NHWC f32
  k_conv3<<<8192, 192, 0, stream>>>(x, wqr, pq_b, xp);
  k_conv3<<<8192, 192, 0, stream>>>(v, wvr, pv_b, vp);
  // LN + window partition -> bf16
  k_ln_win<<<16384, 256, 0, stream>>>(xp, n1w, n1b, xln);
  k_ln_win<<<16384, 256, 0, stream>>>(vp, n2w, n2b, vln);
  // q / k / v projections
  dim3 gg(8192, 3);
  k_gemm<<<gg, 128, 0, stream>>>(xln, C_, qwb, C_, nullptr, qbuf, C_, SCALE_, 0);
  k_gemm<<<gg, 128, 0, stream>>>(vln, C_, kvwb, C_, nullptr, kbuf, C_, 1.f, 1);
  k_gemm<<<gg, 128, 0, stream>>>(vln, C_, kvwb + (size_t)192 * 192, C_, nullptr, vT, C_, 1.f, 2);
  // attention
  k_attn<<<16384, 32, 0, stream>>>(qbuf, kbuf, vT, bF, oa);
  // output projection (f32)
  k_gemm<<<gg, 128, 0, stream>>>(oa, C_, apwb, C_, ap_b, oproj, C_, 1.f, 3);
  // fused MLP + LN4 + residuals -> NCHW output
  k_mlp<<<8192, 256, 0, stream>>>(oproj, n3w, n3b, f1b, fc1_b, f2b, fc2_b,
                                  n4w, n4b, xp, vp, (float*)d_out);
}
